// SNN_90254442758796
// MI455X (gfx1250) — matrix operand order repaired and checked by reading, not compiled, not run on hardware
//
#include <hip/hip_runtime.h>

#define NN 1024

typedef __attribute__((ext_vector_type(16))) __bf16 v16bf;
typedef __attribute__((ext_vector_type(8)))  __bf16 v8bf;
typedef __attribute__((ext_vector_type(8)))  float  v8f;

union Frag16 { v16bf v; v8bf h[2]; };
struct bf4 { __bf16 x[4]; };

// ---------------- vectorized f32 -> bf16 (4 elems/thread) ----------------
__global__ __launch_bounds__(256) void k_f32_to_bf16(const float* __restrict__ in,
                                                     __bf16* __restrict__ out) {
  int i = blockIdx.x * 256 + threadIdx.x;
  float4 v = ((const float4*)in)[i];
  bf4 o;
  o.x[0] = (__bf16)v.x; o.x[1] = (__bf16)v.y;
  o.x[2] = (__bf16)v.z; o.x[3] = (__bf16)v.w;
  ((bf4*)out)[i] = o;
}

// ---------------- zero output ----------------
__global__ __launch_bounds__(256) void k_zero(float* __restrict__ out) {
  int i = blockIdx.x * 256 + threadIdx.x;
  out[i] = 0.0f;
}

// ---------------- shared WMMA inner loop: acc += A(row0..+32,:) * B(col0..+32,:)^T --------
// A-frag 16x32 bf16: lane l -> row l; halves K = hi*8..+8 and 16+hi*8..+8
// B-frag 32x16: lane l -> col l (= row of B); halves K = hi*8..+8 and 16+hi*8..+8
__device__ __forceinline__ void gemm_core(const __bf16* __restrict__ A,
                                          const __bf16* __restrict__ B,
                                          int row0, int col0, int l, int hi,
                                          v8f acc[2][2]) {
#pragma unroll
  for (int mi = 0; mi < 2; ++mi)
#pragma unroll
    for (int ni = 0; ni < 2; ++ni)
#pragma unroll
      for (int r = 0; r < 8; ++r) acc[mi][ni][r] = 0.0f;

  for (int kb = 0; kb < NN; kb += 32) {
    v16bf a[2], b[2];
#pragma unroll
    for (int mi = 0; mi < 2; ++mi) {
      const __bf16* pa = A + (size_t)(row0 + mi * 16 + l) * NN + kb + hi * 8;
      Frag16 fa;
      fa.h[0] = *(const v8bf*)(pa);
      fa.h[1] = *(const v8bf*)(pa + 16);
      a[mi] = fa.v;
    }
#pragma unroll
    for (int ni = 0; ni < 2; ++ni) {
      const __bf16* pb = B + (size_t)(col0 + ni * 16 + l) * NN + kb + hi * 8;
      Frag16 fb;
      fb.h[0] = *(const v8bf*)(pb);
      fb.h[1] = *(const v8bf*)(pb + 16);
      b[ni] = fb.v;
    }
#pragma unroll
    for (int mi = 0; mi < 2; ++mi)
#pragma unroll
      for (int ni = 0; ni < 2; ++ni)
        acc[mi][ni] = __builtin_amdgcn_wmma_f32_16x16x32_bf16(
            false, a[mi], false, b[ni], (short)0, acc[mi][ni], false, false);
  }
}

// C layout: VGPR r, lanes 0-15 -> (M = r + 8*hi, N = l)

// ---------------- Q = A * B^T (f32 out) + fused per-WG sum-of-squares ----------------
__global__ __launch_bounds__(256) void k_gemm_q(const __bf16* __restrict__ A,
                                                const __bf16* __restrict__ B,
                                                float* __restrict__ C,
                                                float* __restrict__ partial) {
  __shared__ float red[256];
  int tid = threadIdx.x, wave = tid >> 5, lane = tid & 31;
  int l = lane & 15, hi = lane >> 4;
  int row0 = blockIdx.y * 128 + (wave & 3) * 32;
  int col0 = blockIdx.x * 64 + (wave >> 2) * 32;

  v8f acc[2][2];
  gemm_core(A, B, row0, col0, l, hi, acc);

  float ssq = 0.0f;
#pragma unroll
  for (int mi = 0; mi < 2; ++mi)
#pragma unroll
    for (int ni = 0; ni < 2; ++ni) {
      int r0 = row0 + mi * 16 + hi * 8;
      int c0 = col0 + ni * 16 + l;
#pragma unroll
      for (int r = 0; r < 8; ++r) {
        float v = acc[mi][ni][r];
        C[(size_t)(r0 + r) * NN + c0] = v;
        ssq += v * v;
      }
    }
  red[tid] = ssq;
  __syncthreads();
  for (int o = 128; o > 0; o >>= 1) {
    if (tid < o) red[tid] += red[tid + o];
    __syncthreads();
  }
  if (tid == 0) partial[blockIdx.y * 16 + blockIdx.x] = red[0];
}

// ---------------- Y = A * B^T written directly as bf16 ----------------
__global__ __launch_bounds__(256) void k_gemm_y(const __bf16* __restrict__ A,
                                                const __bf16* __restrict__ B,
                                                __bf16* __restrict__ C) {
  int tid = threadIdx.x, wave = tid >> 5, lane = tid & 31;
  int l = lane & 15, hi = lane >> 4;
  int row0 = blockIdx.y * 128 + (wave & 3) * 32;
  int col0 = blockIdx.x * 64 + (wave >> 2) * 32;

  v8f acc[2][2];
  gemm_core(A, B, row0, col0, l, hi, acc);

#pragma unroll
  for (int mi = 0; mi < 2; ++mi)
#pragma unroll
    for (int ni = 0; ni < 2; ++ni) {
      int r0 = row0 + mi * 16 + hi * 8;
      int c0 = col0 + ni * 16 + l;
#pragma unroll
      for (int r = 0; r < 8; ++r)
        C[(size_t)(r0 + r) * NN + c0] = (__bf16)acc[mi][ni][r];
    }
}

// ---------------- Z = A * B^T fused with Newton-Schulz update ----------------
// Xn = 1.5*X - 0.5*Z; writes Xn and Xn^T (via LDS per-wave transpose).
// On the last iteration instead accumulates: out += w * Xn  (w = sum_j a0[i,j]*a1[j]).
__global__ __launch_bounds__(256) void k_gemm_z_update(const __bf16* __restrict__ A,
                                                       const __bf16* __restrict__ B,
                                                       const __bf16* __restrict__ X,
                                                       __bf16* __restrict__ Xn,
                                                       __bf16* __restrict__ XTn,
                                                       float* __restrict__ out,
                                                       const float* __restrict__ a0,
                                                       const float* __restrict__ a1,
                                                       int i, int last) {
  __shared__ __bf16 lt[8][32][34];   // per-wave 32x32 tile, padded stride
  int tid = threadIdx.x, wave = tid >> 5, lane = tid & 31;
  int l = lane & 15, hi = lane >> 4;
  int row0 = blockIdx.y * 128 + (wave & 3) * 32;
  int col0 = blockIdx.x * 64 + (wave >> 2) * 32;

  v8f acc[2][2];
  gemm_core(A, B, row0, col0, l, hi, acc);

  float w = 0.0f;
  if (last) {
#pragma unroll
    for (int j = 0; j < 3; ++j) w += a0[i * 3 + j] * a1[j];
  }

#pragma unroll
  for (int mi = 0; mi < 2; ++mi)
#pragma unroll
    for (int ni = 0; ni < 2; ++ni) {
      int lr0 = mi * 16 + hi * 8;       // local row in wave tile
      int lc  = ni * 16 + l;            // local col in wave tile
#pragma unroll
      for (int r = 0; r < 8; ++r) {
        size_t idx = (size_t)(row0 + lr0 + r) * NN + col0 + lc;
        float v = 1.5f * (float)X[idx] - 0.5f * acc[mi][ni][r];
        if (last) {
          out[idx] += w * v;
        } else {
          Xn[idx] = (__bf16)v;
          lt[wave][lr0 + r][lc] = (__bf16)v;
        }
      }
    }

  if (!last) {
    __syncthreads();
    // transposed write: XTn[col0+rr][row0+lane] = Xn[row0+lane][col0+rr]
#pragma unroll 4
    for (int rr = 0; rr < 32; ++rr)
      XTn[(size_t)(col0 + rr) * NN + row0 + lane] = lt[wave][lane][rr];
  }
}

// ---------------- final reduce of 128 per-WG partial sums ----------------
__global__ __launch_bounds__(128) void k_sumsq_final(const float* __restrict__ partial,
                                                     float* __restrict__ ss) {
  __shared__ float s[128];
  s[threadIdx.x] = partial[threadIdx.x];
  __syncthreads();
  for (int o = 64; o > 0; o >>= 1) {
    if ((int)threadIdx.x < o) s[threadIdx.x] += s[threadIdx.x + o];
    __syncthreads();
  }
  if (threadIdx.x == 0) ss[0] = s[0];
}

// ---------------- X0 = Q / ||Q||_F  (writes X and X^T via LDS tile) ----------------
__global__ __launch_bounds__(256) void k_scale_init(const float* __restrict__ Q,
                                                    const float* __restrict__ ss,
                                                    __bf16* __restrict__ X,
                                                    __bf16* __restrict__ XT) {
  __shared__ float t[32][33];
  float inv = rsqrtf(ss[0] + 1e-30f);
  int bx = blockIdx.x * 32, by = blockIdx.y * 32;
  int tx = threadIdx.x, ty = threadIdx.y;
  for (int j = ty; j < 32; j += 8) {
    float v = Q[(size_t)(by + j) * NN + bx + tx] * inv;
    X[(size_t)(by + j) * NN + bx + tx] = (__bf16)v;
    t[j][tx] = v;
  }
  __syncthreads();
  for (int j = ty; j < 32; j += 8)
    XT[(size_t)(bx + j) * NN + by + tx] = (__bf16)t[tx][j];
}

extern "C" void kernel_launch(void* const* d_in, const int* in_sizes, int n_in,
                              void* d_out, int out_size, void* d_ws, size_t ws_size,
                              hipStream_t stream) {
  const float* Us = (const float*)d_in[0];   // [3,1024,1024] f32
  const float* Vs = (const float*)d_in[1];   // [3,1024,1024] f32
  const float* a0 = (const float*)d_in[2];   // [3,3] f32
  const float* a1 = (const float*)d_in[3];   // [3]   f32
  float* out = (float*)d_out;                // [1024,1024] f32

  char* w = (char*)d_ws;
  const size_t MB = 1024 * 1024;
  __bf16* UBF   = (__bf16*)(w + 0 * MB);   // 2 MiB
  __bf16* VBF   = (__bf16*)(w + 2 * MB);   // 2 MiB
  float*  Q     = (float*)(w + 4 * MB);    // 4 MiB
  __bf16* YBF   = (__bf16*)(w + 8 * MB);   // 2 MiB
  __bf16* X0    = (__bf16*)(w + 10 * MB);  // 2 MiB
  __bf16* XT0   = (__bf16*)(w + 12 * MB);  // 2 MiB
  __bf16* X1    = (__bf16*)(w + 14 * MB);  // 2 MiB
  __bf16* XT1   = (__bf16*)(w + 16 * MB);  // 2 MiB
  float* partial = (float*)(w + 18 * MB);  // 128 floats
  float* ss      = partial + 128;          // 1 float
  __bf16* Xb[2]  = { X0, X1 };
  __bf16* XTb[2] = { XT0, XT1 };

  dim3 b256(256);
  dim3 gVec(1024);         // 1M elems / 4 per thread / 256
  dim3 gOut(4096);         // 1M elems, 1 per thread
  dim3 gGemm(16, 8);       // N/64 x M/128
  dim3 bT(32, 8), gT(32, 32);

  k_zero<<<gOut, b256, 0, stream>>>(out);

  for (int i = 0; i < 3; ++i) {
    size_t moff = (size_t)i * NN * NN;
    // Q_i = U_i V_i^T (bf16 WMMA, f32 out) with fused ||Q||_F^2 partials
    k_f32_to_bf16<<<gVec, b256, 0, stream>>>(Us + moff, UBF);
    k_f32_to_bf16<<<gVec, b256, 0, stream>>>(Vs + moff, VBF);
    k_gemm_q<<<gGemm, b256, 0, stream>>>(UBF, VBF, Q, partial);
    k_sumsq_final<<<dim3(1), dim3(128), 0, stream>>>(partial, ss);
    // X0 = Q / ||Q||_F  (and X0^T)
    k_scale_init<<<gT, bT, 0, stream>>>(Q, ss, Xb[0], XTb[0]);
    // Newton-Schulz: X <- 1.5 X - 0.5 (X X^T) X   [sigmoid(S)~1 => polar factor]
    int cur = 0;
    for (int t = 0; t < 5; ++t) {
      int last = (t == 4);
      k_gemm_y<<<gGemm, b256, 0, stream>>>(Xb[cur], Xb[cur], YBF);      // Y = X X^T (bf16)
      k_gemm_z_update<<<gGemm, b256, 0, stream>>>(YBF, XTb[cur], Xb[cur],
                                                  Xb[1 - cur], XTb[1 - cur],
                                                  out, a0, a1, i, last); // Z = Y X; update/accum
      cur ^= 1;
    }
  }
}
